// MultiHeadAttention_37520834298087
// MI455X (gfx1250) — matrix operand order repaired and checked by reading, not compiled, not run on hardware
//
#include <hip/hip_runtime.h>
#include <math.h>

typedef __bf16 bf16;
typedef __attribute__((ext_vector_type(16))) __bf16 v16bf;
typedef __attribute__((ext_vector_type(8)))  __bf16 v8bf;
typedef __attribute__((ext_vector_type(8)))  float  v8f;
typedef __attribute__((ext_vector_type(8)))  unsigned int v8u;

#if defined(__has_builtin)
# if __has_builtin(__builtin_amdgcn_tensor_load_to_lds)
#  define ATTN_USE_TDM 1
# endif
#endif

// ---------------------------------------------------------------------------
// Fragment helpers (wave32 WMMA; contiguous k order k = 16*half + elem, same for A and B)
// ---------------------------------------------------------------------------

// A matrix 16x32 bf16 from row-major storage; `row` -> K=0 of this lane's row.
// lane<16 : elems 0..15 = K0..15
// lane>=16: elems 0..15 = K16..31
__device__ __forceinline__ v16bf load_afrag(const bf16* row, int hh) {
    const bf16* p0 = row + hh * 16;
    v16bf r;
#pragma unroll
    for (int i = 0; i < 16; ++i) { r[i] = p0[i]; }
    return r;
}

// B matrix 32x16 bf16 from B^T storage [n][k] (k contiguous per row).
// lane<16 : elems = K0..15 ; lane>=16: elems = K16..31 (one contiguous run)
__device__ __forceinline__ v16bf load_bfrag(const bf16* row, int hh) {
    const bf16* p = row + hh * 16;
    v16bf r;
#pragma unroll
    for (int i = 0; i < 16; ++i) r[i] = p[i];
    return r;
}

__device__ __forceinline__ v8f wmma_bf16(v16bf a, v16bf b, v8f c) {
    return __builtin_amdgcn_wmma_f32_16x16x32_bf16(
        false, a, false, b, (short)0, c, false, false);
}

__device__ __forceinline__ bf16 cvt_bf16(float x) { return (bf16)x; }
__device__ __forceinline__ bf16 cvt_bf16(bf16 x)  { return x; }
__device__ __forceinline__ void store_out(float* p, float v) { *p = v; }
__device__ __forceinline__ void store_out(bf16*  p, float v) { *p = (bf16)v; }

__device__ __forceinline__ unsigned pack2(float x, float y) {
    unsigned short ux = __builtin_bit_cast(unsigned short, (bf16)x);
    unsigned short uy = __builtin_bit_cast(unsigned short, (bf16)y);
    return (unsigned)ux | ((unsigned)uy << 16);
}

// ---------------------------------------------------------------------------
// Tensor Data Mover: 2D tile (bf16) global -> LDS per CDNA5 ISA ch.8 D#.
// ---------------------------------------------------------------------------
#ifdef ATTN_USE_TDM
typedef __attribute__((ext_vector_type(4))) unsigned int u32x4;
typedef __attribute__((ext_vector_type(4))) int i32x4;
typedef __attribute__((ext_vector_type(8))) int i32x8;

__device__ __forceinline__ void tdm_load2d(unsigned lds_off, const void* gptr,
                                           unsigned width /*elems*/,
                                           unsigned height /*rows*/,
                                           unsigned long long stride /*elems*/) {
    unsigned long long ga = (unsigned long long)(uintptr_t)gptr;
    u32x4 g0;
    g0[0] = 1u;                                   // count=1, user mode
    g0[1] = lds_off;                              // lds_addr (bytes)
    g0[2] = (unsigned)ga;                         // global_addr[31:0]
    g0[3] = (unsigned)((ga >> 32) & 0x01ffffffu)  // global_addr[56:32]
            | (2u << 30);                         // type=2 ("image")
    i32x8 g1;
    g1[0] = (int)(1u << 16);                      // data_size=1 -> 2 bytes
    g1[1] = (int)((width & 0xffffu) << 16);       // tensor_dim0[15:0]
    g1[2] = (int)(((width >> 16) & 0xffffu)       // tensor_dim0[31:16]
            | ((height & 0xffffu) << 16));        // tensor_dim1[15:0]
    g1[3] = (int)(((height >> 16) & 0xffffu)      // tensor_dim1[31:16]
            | ((width & 0xffffu) << 16));         // tile_dim0
    g1[4] = (int)(height & 0xffffu);              // tile_dim1 (tile_dim2=0)
    g1[5] = (int)(unsigned)(stride & 0xffffffffu);        // dim0_stride lo
    g1[6] = (int)(unsigned)((stride >> 32) & 0xffffu);    // dim0_stride hi
    g1[7] = 0;                                    // dim1_stride (2D: unused)
    i32x4 z4 = {0, 0, 0, 0};
#if defined(__clang_major__) && (__clang_major__ >= 23)
    i32x8 z8 = {0, 0, 0, 0, 0, 0, 0, 0};
    __builtin_amdgcn_tensor_load_to_lds(g0, g1, z4, z4, z8, 0);
#else
    __builtin_amdgcn_tensor_load_to_lds(g0, g1, z4, z4, 0);
#endif
}
#endif

// ---------------------------------------------------------------------------
// GEMM:  Y = (A[M,K] @ W[K,N] + bias) * out_scale
// TRANSPOSE_V=false: Y[M,N] row-major (OT f32 or bf16)
// TRANSPOSE_V=true : Y stored as Yt[b][N][Sdim] bf16 (per-batch transpose);
//                    lanes own columns -> contiguous 16B vector stores.
// Block: 128 threads (4 waves), 128x64 output tile; wave w -> 32 rows.
// ---------------------------------------------------------------------------
template <typename AT, typename OT, bool TRANSPOSE_V>
__global__ __launch_bounds__(128) void gemm_bias_wmma(
    const AT* __restrict__ A, const float* __restrict__ W,
    const float* __restrict__ bias, OT* __restrict__ Y,
    int M, int N, int K, int Sdim, float out_scale) {
    __shared__ bf16 As[128][32];   // [m][k]  8 KB
    __shared__ bf16 Bs[64][32];    // [n][k]  4 KB (W tile transposed)

    const int tid  = threadIdx.x;
    const int wave = tid >> 5;
    const int lane = tid & 31;
    const int l    = lane & 15;
    const int hh   = lane >> 4;
    const int m0   = blockIdx.x * 128;
    const int n0   = blockIdx.y * 64;

    v8f acc[2][4] = {{v8f{}, v8f{}, v8f{}, v8f{}},
                     {v8f{}, v8f{}, v8f{}, v8f{}}};

    for (int k0 = 0; k0 < K; k0 += 32) {
        __syncthreads();
        for (int idx = tid; idx < 128 * 32; idx += 128) {     // coalesced in k
            int r = idx >> 5, c = idx & 31;
            As[r][c] = cvt_bf16(A[(size_t)(m0 + r) * K + (k0 + c)]);
        }
        for (int idx = tid; idx < 64 * 32; idx += 128) {      // coalesced in n
            int kk = idx >> 6, n = idx & 63;
            Bs[n][kk] = (bf16)W[(size_t)(k0 + kk) * N + (n0 + n)];
        }
        __syncthreads();

        v16bf af0 = load_afrag(&As[wave * 32 + l][0], hh);
        v16bf af1 = load_afrag(&As[wave * 32 + 16 + l][0], hh);
#pragma unroll
        for (int j = 0; j < 4; ++j) {
            v16bf bfb = load_bfrag(&Bs[j * 16 + l][0], hh);
            acc[0][j] = wmma_bf16(af0, bfb, acc[0][j]);
            acc[1][j] = wmma_bf16(af1, bfb, acc[1][j]);
        }
    }

    if constexpr (TRANSPOSE_V) {
        const int bb    = m0 / Sdim;
        const int sbase = (m0 % Sdim) + wave * 32;
#pragma unroll
        for (int j = 0; j < 4; ++j) {
            int col = n0 + j * 16 + l;
            float bv = bias[col];
            bf16* colbase = (bf16*)Y + ((size_t)bb * N + col) * Sdim;
#pragma unroll
            for (int i = 0; i < 2; ++i) {
                v8bf pk;
#pragma unroll
                for (int r = 0; r < 8; ++r)
                    pk[r] = (bf16)((acc[i][j][r] + bv) * out_scale);
                *(v8bf*)&colbase[sbase + i * 16 + 8 * hh] = pk;
            }
        }
    } else {
#pragma unroll
        for (int j = 0; j < 4; ++j) {
            int col = n0 + j * 16 + l;
            float bv = bias[col];
#pragma unroll
            for (int i = 0; i < 2; ++i)
#pragma unroll
                for (int r = 0; r < 8; ++r) {
                    int row = m0 + wave * 32 + i * 16 + r + 8 * hh;
                    store_out(&Y[(size_t)row * N + col],
                              (acc[i][j][r] + bv) * out_scale);
                }
        }
    }
}

// ---------------------------------------------------------------------------
// Flash attention, transposed formulation: S^T = K·Q^T, O^T = V^T·P^T.
// Q carries 0.125*log2(e) (exp2-domain softmax). V arrives pre-transposed
// (Vt[b][d_model][S]) so both K and V tiles are plain 2D tiles staged by the
// Tensor Data Mover with double buffering (DMA for chunk i+1 overlaps
// compute on chunk i; one barrier per chunk).
// Block: 128 threads = 4 waves; block = 64 queries of one (b,h);
// wave owns a 16-query tile; keys streamed in 64-key chunks.
// ---------------------------------------------------------------------------
__global__ __launch_bounds__(128) void attn_fa_wmma(
    const bf16* __restrict__ Qp, const bf16* __restrict__ Kp,
    const bf16* __restrict__ Vt, bf16* __restrict__ Ao,
    int S, int H, int D) {
    __shared__ __align__(16) bf16 Ks[2][64][64];   // key rows [k][d]  16 KB
    __shared__ __align__(16) bf16 Vs[2][64][64];   // V^T rows [d][k]  16 KB

    const int tid  = threadIdx.x;
    const int wave = tid >> 5;
    const int lane = tid & 31;
    const int l    = lane & 15;
    const int hh   = lane >> 4;

    const int nqt = S / 64;
    const int qt  = blockIdx.x % nqt;
    const int h   = (blockIdx.x / nqt) % H;
    const int b   = blockIdx.x / (nqt * H);
    const int q0  = qt * 64 + wave * 16;

    const size_t rs = (size_t)H * 64;   // QKV row stride in elements

    // Q as B-fragments (Q^T): lane = query column, contiguous d per half.
    const bf16* qrow = Qp + ((size_t)(b * S) + q0 + l) * rs + (size_t)h * 64;
    v16bf qb0 = load_bfrag(qrow, hh);        // d 0..31
    v16bf qb1 = load_bfrag(qrow + 32, hh);   // d 32..63

    v8f o[4] = {v8f{}, v8f{}, v8f{}, v8f{}}; // O^T tiles: d-chunks x 16 queries
    float m = -3.0e38f, lsum = 0.0f;

    const bf16* Kbase = Kp + (size_t)(b * S) * rs + (size_t)h * 64;
    const bf16* Vtb   = Vt + ((size_t)b * D + (size_t)h * 64) * S;

#ifdef ATTN_USE_TDM
    if (wave == 0) {
        tdm_load2d((unsigned)(uintptr_t)&Ks[0][0][0], Kbase, 64, 64, rs);
        tdm_load2d((unsigned)(uintptr_t)&Vs[0][0][0], Vtb, 64, 64,
                   (unsigned long long)S);
    }
#else
    for (int idx = tid; idx < 64 * 64; idx += 128) {
        int r = idx >> 6, c = idx & 63;
        Ks[0][r][c] = Kbase[(size_t)r * rs + c];
        Vs[0][r][c] = Vtb[(size_t)r * S + c];
    }
#endif

    for (int kc = 0; kc < S; kc += 64) {
        const int cur = (kc >> 6) & 1;
#ifdef ATTN_USE_TDM
        if (wave == 0) __builtin_amdgcn_s_wait_tensorcnt(0);
        __syncthreads();                       // tile ready; prev buffer free
        if (wave == 0 && kc + 64 < S) {        // prefetch next chunk via TDM
            tdm_load2d((unsigned)(uintptr_t)&Ks[cur ^ 1][0][0],
                       Kbase + (size_t)(kc + 64) * rs, 64, 64, rs);
            tdm_load2d((unsigned)(uintptr_t)&Vs[cur ^ 1][0][0],
                       Vtb + (kc + 64), 64, 64, (unsigned long long)S);
        }
#else
        __syncthreads();
#endif

        // S^T tiles: 4 x (16 keys x 16 queries), contraction over d = 64
        v8f s[4];
#pragma unroll
        for (int t = 0; t < 4; ++t) {
            const bf16* krow = &Ks[cur][t * 16 + l][0];
            v8f z = {};
            v8f p = wmma_bf16(load_afrag(krow, hh), qb0, z);
            s[t]  = wmma_bf16(load_afrag(krow + 32, hh), qb1, p);
        }

        // online softmax in exp2 domain; per-lane = per-query stats
        float mx = -3.0e38f;
#pragma unroll
        for (int t = 0; t < 4; ++t)
#pragma unroll
            for (int r = 0; r < 8; ++r) mx = fmaxf(mx, s[t][r]);
        mx = fmaxf(mx, __shfl_xor(mx, 16, 32));

        float nm    = fmaxf(m, mx);
        float alpha = exp2f(m - nm);
        float sum   = 0.0f;
#pragma unroll
        for (int t = 0; t < 4; ++t)
#pragma unroll
            for (int r = 0; r < 8; ++r) {
                float p = exp2f(s[t][r] - nm);
                s[t][r] = p;
                sum += p;
            }
        sum += __shfl_xor(sum, 16, 32);
        lsum = lsum * alpha + sum;
        m = nm;
#pragma unroll
        for (int j = 0; j < 4; ++j)
#pragma unroll
            for (int r = 0; r < 8; ++r) o[j][r] *= alpha;

        // P^T (C layout) -> B-fragment layout via register exchange
        v16bf pf[2];
#pragma unroll
        for (int g = 0; g < 2; ++g) {
            unsigned a[4], bp[4];
#pragma unroll
            for (int i = 0; i < 4; ++i) {
                a[i]  = pack2(s[2 * g][2 * i],     s[2 * g][2 * i + 1]);
                bp[i] = pack2(s[2 * g + 1][2 * i], s[2 * g + 1][2 * i + 1]);
            }
            v8u w;
#pragma unroll
            for (int i = 0; i < 4; ++i) {
                unsigned ax = __shfl_xor(a[i], 16, 32);
                unsigned bx = __shfl_xor(bp[i], 16, 32);
                w[i]     = hh ? bx    : a[i];   // keys base+0..7  / +16..23
                w[i + 4] = hh ? bp[i] : ax;     // keys base+8..15 / +24..31
            }
            pf[g] = __builtin_bit_cast(v16bf, w);
        }

        // O^T += V^T · P^T : A-frags from pre-transposed V tile, K-dim = 64
#pragma unroll
        for (int j = 0; j < 4; ++j) {
            const bf16* vrow = &Vs[cur][j * 16 + l][0];
            o[j] = wmma_bf16(load_afrag(vrow, hh),      pf[0], o[j]);
            o[j] = wmma_bf16(load_afrag(vrow + 32, hh), pf[1], o[j]);
        }

#ifndef ATTN_USE_TDM
        __syncthreads();
        if (kc + 64 < S) {
            for (int idx = tid; idx < 64 * 64; idx += 128) {
                int r = idx >> 6, c = idx & 63;
                Ks[cur ^ 1][r][c] = Kbase[(size_t)(kc + 64 + r) * rs + c];
                Vs[cur ^ 1][r][c] = Vtb[(size_t)r * S + kc + 64 + c];
            }
        }
#endif
    }

    // normalize and store: lane pair shares one query row -> vector stores
    float inv = 1.0f / lsum;
    bf16* orow = Ao + ((size_t)(b * S) + q0 + l) * rs + (size_t)h * 64;
#pragma unroll
    for (int j = 0; j < 4; ++j) {
        v8bf pk;
#pragma unroll
        for (int r = 0; r < 8; ++r) pk[r] = (bf16)(o[j][r] * inv);
        *(v8bf*)&orow[j * 16 + 8 * hh] = pk;   // d = j*16 + 8*hh + r
    }
}

// ---------------------------------------------------------------------------
// Launcher
// ---------------------------------------------------------------------------
extern "C" void kernel_launch(void* const* d_in, const int* in_sizes, int n_in,
                              void* d_out, int out_size, void* d_ws, size_t ws_size,
                              hipStream_t stream) {
    const float* query  = (const float*)d_in[0];
    const float* keys   = (const float*)d_in[1];
    const float* values = (const float*)d_in[2];
    const float* Wq = (const float*)d_in[3];
    const float* bq = (const float*)d_in[4];
    const float* Wk = (const float*)d_in[5];
    const float* bk = (const float*)d_in[6];
    const float* Wv = (const float*)d_in[7];
    const float* bv = (const float*)d_in[8];
    const float* Wo = (const float*)d_in[9];
    const float* bo = (const float*)d_in[10];

    const int B = 2, S = 2048, D = 1024, H = 16;
    const int M = B * S;                 // 4096 rows

    bf16* Qp = (bf16*)d_ws;              // [M, D] bf16
    bf16* Kp = Qp + (size_t)M * D;       // [M, D] bf16
    bf16* Vt = Kp + (size_t)M * D;       // [B, D, S] bf16 (pre-transposed V)
    bf16* Aout = Vt + (size_t)M * D;     // attention output [M, D] bf16

    dim3 gg(M / 128, D / 64);            // (32, 16)

    // Q carries 1/sqrt(d_head) * log2(e) for exp2-domain softmax
    const float qscale = 0.125f * 1.4426950408889634f;
    gemm_bias_wmma<float, bf16, false><<<gg, 128, 0, stream>>>(
        query,  Wq, bq, Qp, M, D, D, S, qscale);
    gemm_bias_wmma<float, bf16, false><<<gg, 128, 0, stream>>>(
        keys,   Wk, bk, Kp, M, D, D, S, 1.0f);
    gemm_bias_wmma<float, bf16, true><<<gg, 128, 0, stream>>>(
        values, Wv, bv, Vt, M, D, D, S, 1.0f);

    attn_fa_wmma<<<dim3(B * H * (S / 64)), 128, 0, stream>>>(
        Qp, Kp, Vt, Aout, S, H, D);

    gemm_bias_wmma<bf16, float, false><<<gg, 128, 0, stream>>>(
        Aout, Wo, bo, (float*)d_out, M, D, D, S, 1.0f);
}
